// CNNModel_46282567582167
// MI455X (gfx1250) — matrix operand order repaired and checked by reading, not compiled, not run on hardware
//
#include <hip/hip_runtime.h>
#include <hip/hip_bf16.h>
#include <math.h>

// ---------------------------------------------------------------------------
// WMMA helpers (CDNA5 / gfx1250, wave32)
// ---------------------------------------------------------------------------
typedef _Float16 v16h_t __attribute__((ext_vector_type(16)));
typedef _Float16 v8h_t  __attribute__((ext_vector_type(8)));
typedef float    v8f_t  __attribute__((ext_vector_type(8)));

// A fragment: 16x32 f16 tile, row-major source, row stride = lda.
// lane: row = lane&15, kbase = (lane>>4)*16 ; elements 0..15 = K[kb..kb+15]
// (contiguous k order, matching the B fragment).
__device__ __forceinline__ v16h_t load_frag_a(const _Float16* p) {
  v8h_t lo = *reinterpret_cast<const v8h_t*>(p);
  v8h_t hi = *reinterpret_cast<const v8h_t*>(p + 8);
  v16h_t r;
#pragma unroll
  for (int i = 0; i < 8; ++i) { r[i] = lo[i]; r[i + 8] = hi[i]; }
  return r;
}

// B fragment: 32x16 f16 tile where B[k][n] = W[n][k] (W row-major N x K).
// lane: n = lane&15, kbase = (lane>>4)*16 ; elements 0..15 = W[n][kbase+0..15]
// -> 32 contiguous bytes.
__device__ __forceinline__ v16h_t load_frag_b(const _Float16* p) {
  v8h_t lo = *reinterpret_cast<const v8h_t*>(p);
  v8h_t hi = *reinterpret_cast<const v8h_t*>(p + 8);
  v16h_t r;
#pragma unroll
  for (int i = 0; i < 8; ++i) { r[i] = lo[i]; r[i + 8] = hi[i]; }
  return r;
}

#define WMMA_F16(a, b, c) \
  __builtin_amdgcn_wmma_f32_16x16x32_f16(false, (a), false, (b), (short)0, (c), false, false)

static constexpr int kB = 32;     // batch
static constexpr int kT = 1024;   // time steps
static constexpr int kH = 256;    // GRU hidden per direction
static constexpr float kEPS = 1e-5f;

// ---------------------------------------------------------------------------
// CNN kernels (plain VALU; ~16 GFLOP total, massively parallel)
// ---------------------------------------------------------------------------
__global__ void conv1_bn_relu(const float* __restrict__ in, const float* __restrict__ w,
                              const float* __restrict__ bias, const float* __restrict__ bs,
                              const float* __restrict__ bb, const float* __restrict__ bm,
                              const float* __restrict__ bv, float* __restrict__ out, int total) {
  int idx = blockIdx.x * blockDim.x + threadIdx.x;
  if (idx >= total) return;
  int ow = idx % 13; int t = idx / 13;
  int oh = t % 765;  t /= 765;
  int oc = t % 32;   int b_ = t / 32;
  const float* ip = in + (size_t)b_ * 196608 + (size_t)(oh * 4) * 64 + ow * 4;
  const float* wp = w + oc * 256;
  float s = 0.f;
  for (int kh = 0; kh < 16; ++kh)
#pragma unroll
    for (int kw = 0; kw < 16; ++kw) s += ip[kh * 64 + kw] * wp[kh * 16 + kw];
  float sc = bs[oc] * rsqrtf(bv[oc] + kEPS);
  float y = (s + bias[oc] - bm[oc]) * sc + bb[oc];
  out[idx] = fmaxf(y, 0.f);
}

__global__ void conv2_bn_relu_pool(const float* __restrict__ a1, const float* __restrict__ w,
                                   const float* __restrict__ bias, const float* __restrict__ bs,
                                   const float* __restrict__ bb, const float* __restrict__ bm,
                                   const float* __restrict__ bv, float* __restrict__ out, int total) {
  int idx = blockIdx.x * blockDim.x + threadIdx.x;
  if (idx >= total) return;
  int pw = idx % 5;  int t = idx / 5;
  int ph = t % 381;  t /= 381;
  int oc = t % 64;   int b_ = t / 64;
  float sc = bs[oc] * rsqrtf(bv[oc] + kEPS);
  float add = (bias[oc] - bm[oc]) * sc + bb[oc];
  float best = 0.f;  // relu() >= 0 always
#pragma unroll
  for (int dh = 0; dh < 2; ++dh)
#pragma unroll
    for (int dw = 0; dw < 2; ++dw) {
      int oh = 2 * ph + dh, ow = 2 * pw + dw;
      float s = 0.f;
      for (int ic = 0; ic < 32; ++ic) {
        const float* ap = a1 + ((size_t)(b_ * 32 + ic) * 765 + oh) * 13 + ow;
        const float* wp = w + (oc * 32 + ic) * 9;
#pragma unroll
        for (int kh = 0; kh < 3; ++kh)
#pragma unroll
          for (int kw = 0; kw < 3; ++kw) s += ap[kh * 13 + kw] * wp[kh * 3 + kw];
      }
      best = fmaxf(best, fmaxf(s * sc + add, 0.f));
    }
  out[idx] = best;
}

__global__ void conv3_relu(const float* __restrict__ a2, const float* __restrict__ w,
                           const float* __restrict__ bias, float* __restrict__ out, int total) {
  int idx = blockIdx.x * blockDim.x + threadIdx.x;
  if (idx >= total) return;
  int ow = idx % 4;  int t = idx / 4;
  int oh = t % 189;  t /= 189;
  int oc = t % 8;    int b_ = t / 8;
  float s = 0.f;
  for (int ic = 0; ic < 64; ++ic) {
    const float* ap = a2 + ((size_t)(b_ * 64 + ic) * 381 + 2 * oh) * 5 + ow;
    const float* wp = w + (oc * 64 + ic) * 8;
#pragma unroll
    for (int kh = 0; kh < 4; ++kh)
#pragma unroll
      for (int kw = 0; kw < 2; ++kw) s += ap[kh * 5 + kw] * wp[kh * 2 + kw];
  }
  out[idx] = fmaxf(s + bias[oc], 0.f);
}

// conv4 + relu + bilinear resize (186,3)->(80,3), one block per batch element
__global__ void conv4_resize(const float* __restrict__ a3, const float* __restrict__ w4,
                             const float* __restrict__ b4, float* __restrict__ ssum) {
  __shared__ float c4[186 * 3];
  const int b_ = blockIdx.x;
  const int tid = threadIdx.x;
  for (int i = tid; i < 186 * 3; i += blockDim.x) {
    int oh = i / 3, ow = i % 3;
    float s = 0.f;
#pragma unroll
    for (int ic = 0; ic < 8; ++ic) {
      const float* ap = a3 + ((size_t)(b_ * 8 + ic) * 189 + oh) * 4 + ow;
      const float* wp = w4 + ic * 8;
#pragma unroll
      for (int kh = 0; kh < 4; ++kh)
#pragma unroll
        for (int kw = 0; kw < 2; ++kw) s += ap[kh * 4 + kw] * wp[kh * 2 + kw];
    }
    c4[i] = fmaxf(s + b4[0], 0.f);
  }
  __syncthreads();
  const float scale = 186.0f / 80.0f;  // half-pixel-center linear resize
  for (int i = tid; i < 240; i += blockDim.x) {
    int oi = i / 3, ow = i % 3;
    float src = (oi + 0.5f) * scale - 0.5f;
    float fl = floorf(src);
    int i0 = (int)fl;
    float f = src - fl;
    if (i0 < 0) { i0 = 0; f = 0.f; }
    if (i0 > 185) i0 = 185;
    int i1 = i0 + 1; if (i1 > 185) i1 = 185;
    ssum[b_ * 240 + i] = c4[i0 * 3 + ow] * (1.f - f) + c4[i1 * 3 + ow] * f;
  }
}

// ---------------------------------------------------------------------------
// Data prep kernels
// ---------------------------------------------------------------------------
// x (T,B,448) f16: [0,192)=raw features, [192,432)=summary, [432,448)=zero pad
__global__ void build_x_f16(const float* __restrict__ batch, const float* __restrict__ ssum,
                            _Float16* __restrict__ x, int total) {
  int idx = blockIdx.x * blockDim.x + threadIdx.x;
  if (idx >= total) return;
  int k = idx % 448;
  int m = idx / 448;           // m = t*B + b
  int b_ = m & 31;
  int t = m >> 5;
  float v;
  if (k < 192)       v = batch[((size_t)b_ * kT + t) * 192 + k];
  else if (k < 432)  v = ssum[b_ * 240 + (k - 192)];
  else               v = 0.f;
  x[idx] = (_Float16)v;
}

// f32 -> f16 with zero padding (rows and cols)
__global__ void cvt_f16_pad(const float* __restrict__ src, _Float16* __restrict__ dst,
                            int srcRows, int srcCols, int dstStride, int total) {
  int idx = blockIdx.x * blockDim.x + threadIdx.x;
  if (idx >= total) return;
  int r = idx / dstStride, c = idx % dstStride;
  dst[idx] = (r < srcRows && c < srcCols) ? (_Float16)src[r * srcCols + c] : (_Float16)0.f;
}

// ---------------------------------------------------------------------------
// WMMA GEMM: Y(M,N) f32 = A(M,Kp) f16 @ W(N,Kp)^T f16 + bias(N)
// block = 256 (8 waves): 128-row x 64-col tile; wave = 16x64 (4 accumulators)
// ---------------------------------------------------------------------------
__global__ __launch_bounds__(256) void gemm_wmma_bias(
    const _Float16* __restrict__ A, const _Float16* __restrict__ W,
    const float* __restrict__ bias, float* __restrict__ Y, int N, int Kp) {
  const int lane = threadIdx.x & 31;
  const int wave = threadIdx.x >> 5;
  const int m0 = blockIdx.y * 128 + wave * 16;
  const int n0 = blockIdx.x * 64;
  const int arow = lane & 15, ahk = lane >> 4;
  const int bn = lane & 15, bkb = (lane >> 4) * 16;
  const _Float16* ap = A + (size_t)(m0 + arow) * Kp + ahk * 16;
  v8f_t acc[4] = {v8f_t{}, v8f_t{}, v8f_t{}, v8f_t{}};
  for (int k0 = 0; k0 < Kp; k0 += 32) {
    v16h_t a = load_frag_a(ap + k0);
    const _Float16* wp = W + (size_t)(n0 + bn) * Kp + k0 + bkb;
    v16h_t b0 = load_frag_b(wp);
    v16h_t b1 = load_frag_b(wp + (size_t)16 * Kp);
    v16h_t b2 = load_frag_b(wp + (size_t)32 * Kp);
    v16h_t b3 = load_frag_b(wp + (size_t)48 * Kp);
    acc[0] = WMMA_F16(a, b0, acc[0]);
    acc[1] = WMMA_F16(a, b1, acc[1]);
    acc[2] = WMMA_F16(a, b2, acc[2]);
    acc[3] = WMMA_F16(a, b3, acc[3]);
  }
  const int cn = lane & 15, cmb = (lane >> 4) * 8;
#pragma unroll
  for (int j = 0; j < 4; ++j) {
    int col = n0 + j * 16 + cn;
    float bvv = bias[col];
#pragma unroll
    for (int i = 0; i < 8; ++i)
      Y[(size_t)(m0 + cmb + i) * N + col] = acc[j][i] + bvv;
  }
}

// ---------------------------------------------------------------------------
// Persistent GRU recurrence. grid=2 (blockIdx 0=forward, 1=reverse),
// block=1024 (32 waves). h lives in LDS (f16); per step:
//   hg(32,768) = h(32,256) @ w_hh(768,256)^T  via 96 16x16 WMMA tiles,
//   then fused gate math, h update, f16 output store.
// ---------------------------------------------------------------------------
__global__ __launch_bounds__(1024) void gru_layer(
    const float* __restrict__ xgF, const float* __restrict__ xgR,
    const _Float16* __restrict__ whhF, const _Float16* __restrict__ whhR,
    const float* __restrict__ bhhF, const float* __restrict__ bhhR,
    _Float16* __restrict__ out, int T) {
  __shared__ __align__(16) _Float16 h_sh[kB * kH];       // 16 KB
  __shared__ __align__(16) float hg_sh[kB * 3 * kH];     // 96 KB
  __shared__ float bhh_sh[3 * kH];                       // 3 KB
  const int rev = blockIdx.x;
  const float* xg = rev ? xgR : xgF;
  const _Float16* whh = rev ? whhR : whhF;
  const float* bhh = rev ? bhhR : bhhF;
  const int outOff = rev ? kH : 0;
  const int tid = threadIdx.x;

  for (int i = tid; i < kB * kH; i += 1024) h_sh[i] = (_Float16)0.f;
  for (int i = tid; i < 3 * kH; i += 1024) bhh_sh[i] = bhh[i];
  __syncthreads();

  const int lane = tid & 31, wave = tid >> 5;
  const int arow = lane & 15, ahk = lane >> 4;
  const int bn = lane & 15, bkb = (lane >> 4) * 16;
  const int cn = lane & 15, cmb = (lane >> 4) * 8;

  for (int s = 0; s < T; ++s) {
    const int t = rev ? (T - 1 - s) : s;
    // ---- recurrent matmul: 96 tiles (2 M-tiles x 48 N-tiles), 3 per wave
#pragma unroll
    for (int q = 0; q < 3; ++q) {
      const int tile = wave * 3 + q;
      const int m0 = (tile & 1) * 16;
      const int n0 = (tile >> 1) * 16;
      v8f_t acc = {};
      const _Float16* ap = h_sh + (m0 + arow) * kH + ahk * 16;
      const _Float16* wp = whh + (size_t)(n0 + bn) * kH + bkb;
#pragma unroll
      for (int k0 = 0; k0 < kH; k0 += 32) {
        v16h_t a = load_frag_a(ap + k0);
        v16h_t b = load_frag_b(wp + k0);
        acc = WMMA_F16(a, b, acc);
      }
#pragma unroll
      for (int i = 0; i < 8; ++i)
        hg_sh[(m0 + cmb + i) * (3 * kH) + n0 + cn] = acc[i];
    }
    __syncthreads();
    // ---- fused gates: r,z,n -> h_new
    const float* xrow = xg + (size_t)t * kB * (3 * kH);
    for (int i = tid; i < kB * kH; i += 1024) {
      const int b_ = i >> 8;
      const int j = i & (kH - 1);
      const float* xb = xrow + b_ * (3 * kH);
      const float* hb = hg_sh + b_ * (3 * kH);
      float r = 1.f / (1.f + expf(-(xb[j] + hb[j] + bhh_sh[j])));
      float z = 1.f / (1.f + expf(-(xb[kH + j] + hb[kH + j] + bhh_sh[kH + j])));
      float n = tanhf(xb[2 * kH + j] + r * (hb[2 * kH + j] + bhh_sh[2 * kH + j]));
      float hold = (float)h_sh[i];
      float hnew = (1.f - z) * n + z * hold;
      h_sh[i] = (_Float16)hnew;
      out[((size_t)t * kB + b_) * (2 * kH) + outOff + j] = (_Float16)hnew;
    }
    __syncthreads();
  }
}

// ---------------------------------------------------------------------------
// FC + length mask: out(b*T+t, 62) = mask(t<len[b]) * (H(t*B+b,:) @ fc_w^T + b)
// H: (T*B, 512) f16 ; fc weights padded to (64,512) f16.
// ---------------------------------------------------------------------------
__global__ __launch_bounds__(256) void fc_mask(
    const _Float16* __restrict__ H, const _Float16* __restrict__ Wf,
    const float* __restrict__ bias, const int* __restrict__ lengths,
    float* __restrict__ out, int T) {
  const int lane = threadIdx.x & 31;
  const int wave = threadIdx.x >> 5;
  const int m0 = blockIdx.x * 128 + wave * 16;
  const int arow = lane & 15, ahk = lane >> 4;
  const int bn = lane & 15, bkb = (lane >> 4) * 16;
  v8f_t acc[4] = {v8f_t{}, v8f_t{}, v8f_t{}, v8f_t{}};
  const _Float16* ap = H + (size_t)(m0 + arow) * 512 + ahk * 16;
  for (int k0 = 0; k0 < 512; k0 += 32) {
    v16h_t a = load_frag_a(ap + k0);
#pragma unroll
    for (int j = 0; j < 4; ++j) {
      v16h_t b = load_frag_b(Wf + (size_t)(j * 16 + bn) * 512 + k0 + bkb);
      acc[j] = WMMA_F16(a, b, acc[j]);
    }
  }
  const int cn = lane & 15, cmb = (lane >> 4) * 8;
#pragma unroll
  for (int j = 0; j < 4; ++j) {
    int col = j * 16 + cn;
    if (col >= 62) continue;
    float bvv = bias[col];
#pragma unroll
    for (int i = 0; i < 8; ++i) {
      int m = m0 + cmb + i;   // m = t*B + b
      int t = m >> 5;
      int b_ = m & 31;
      float v = (t < lengths[b_]) ? (acc[j][i] + bvv) : 0.f;
      out[((size_t)b_ * T + t) * 62 + col] = v;
    }
  }
}

// ---------------------------------------------------------------------------
// Host launcher
// ---------------------------------------------------------------------------
extern "C" void kernel_launch(void* const* d_in, const int* in_sizes, int n_in,
                              void* d_out, int out_size, void* d_ws, size_t ws_size,
                              hipStream_t stream) {
  (void)in_sizes; (void)n_in; (void)out_size; (void)ws_size;
  const float* batch = (const float*)d_in[0];
  const int* lengths = (const int*)d_in[1];
  const float* c1_w = (const float*)d_in[2];  const float* c1_b = (const float*)d_in[3];
  const float* bn1_s = (const float*)d_in[4]; const float* bn1_b = (const float*)d_in[5];
  const float* bn1_m = (const float*)d_in[6]; const float* bn1_v = (const float*)d_in[7];
  const float* c2_w = (const float*)d_in[8];  const float* c2_b = (const float*)d_in[9];
  const float* bn2_s = (const float*)d_in[10]; const float* bn2_b = (const float*)d_in[11];
  const float* bn2_m = (const float*)d_in[12]; const float* bn2_v = (const float*)d_in[13];
  const float* c3_w = (const float*)d_in[14]; const float* c3_b = (const float*)d_in[15];
  const float* c4_w = (const float*)d_in[16]; const float* c4_b = (const float*)d_in[17];
  const float* w_ih_l0 = (const float*)d_in[18]; const float* w_hh_l0 = (const float*)d_in[19];
  const float* b_ih_l0 = (const float*)d_in[20]; const float* b_hh_l0 = (const float*)d_in[21];
  const float* w_ih_l0r = (const float*)d_in[22]; const float* w_hh_l0r = (const float*)d_in[23];
  const float* b_ih_l0r = (const float*)d_in[24]; const float* b_hh_l0r = (const float*)d_in[25];
  const float* w_ih_l1 = (const float*)d_in[26]; const float* w_hh_l1 = (const float*)d_in[27];
  const float* b_ih_l1 = (const float*)d_in[28]; const float* b_hh_l1 = (const float*)d_in[29];
  const float* w_ih_l1r = (const float*)d_in[30]; const float* w_hh_l1r = (const float*)d_in[31];
  const float* b_ih_l1r = (const float*)d_in[32]; const float* b_hh_l1r = (const float*)d_in[33];
  const float* fc_w = (const float*)d_in[34]; const float* fc_b = (const float*)d_in[35];
  float* out = (float*)d_out;

  const int M = kT * kB;  // 32768 GEMM rows (m = t*B + b)

  // ---- workspace layout
  size_t off = 0;
  auto wsa = [&](size_t bytes) -> char* {
    char* p = (char*)d_ws + off;
    off += (bytes + 255) & ~(size_t)255;
    return p;
  };
  _Float16* xf16  = (_Float16*)wsa((size_t)M * 448 * 2);  // padded K=448
  _Float16* h0f16 = (_Float16*)wsa((size_t)M * 512 * 2);
  _Float16* h1f16 = (_Float16*)wsa((size_t)M * 512 * 2);
  float* xgF = (float*)wsa((size_t)M * 768 * 4);
  float* xgR = (float*)wsa((size_t)M * 768 * 4);
  _Float16* wih0f  = (_Float16*)wsa((size_t)768 * 448 * 2);
  _Float16* wih0rf = (_Float16*)wsa((size_t)768 * 448 * 2);
  _Float16* whh0f  = (_Float16*)wsa((size_t)768 * 256 * 2);
  _Float16* whh0rf = (_Float16*)wsa((size_t)768 * 256 * 2);
  _Float16* wih1f  = (_Float16*)wsa((size_t)768 * 512 * 2);
  _Float16* wih1rf = (_Float16*)wsa((size_t)768 * 512 * 2);
  _Float16* whh1f  = (_Float16*)wsa((size_t)768 * 256 * 2);
  _Float16* whh1rf = (_Float16*)wsa((size_t)768 * 256 * 2);
  _Float16* fcwf   = (_Float16*)wsa((size_t)64 * 512 * 2);
  float* ssum = (float*)wsa((size_t)kB * 240 * 4);
  // CNN scratch aliases the (not-yet-live) xg regions
  float* a1 = xgF;                                   // (32,32,765,13)  ~40.7 MB
  float* a2 = xgR;                                   // (32,64,381,5)   ~15.6 MB
  float* a3 = (float*)((char*)xgR + (size_t)64 * 1024 * 1024);  // (32,8,189,4)

  // ---- CNN summary path
  {
    int total = kB * 32 * 765 * 13;
    conv1_bn_relu<<<(total + 255) / 256, 256, 0, stream>>>(
        batch, c1_w, c1_b, bn1_s, bn1_b, bn1_m, bn1_v, a1, total);
  }
  {
    int total = kB * 64 * 381 * 5;
    conv2_bn_relu_pool<<<(total + 255) / 256, 256, 0, stream>>>(
        a1, c2_w, c2_b, bn2_s, bn2_b, bn2_m, bn2_v, a2, total);
  }
  {
    int total = kB * 8 * 189 * 4;
    conv3_relu<<<(total + 255) / 256, 256, 0, stream>>>(a2, c3_w, c3_b, a3, total);
  }
  conv4_resize<<<kB, 256, 0, stream>>>(a3, c4_w, c4_b, ssum);

  // ---- build f16 inputs / weights
  {
    int total = M * 448;
    build_x_f16<<<(total + 255) / 256, 256, 0, stream>>>(batch, ssum, xf16, total);
  }
  auto cvt = [&](const float* src, _Float16* dst, int sr, int sc, int stride, int dr) {
    int total = dr * stride;
    cvt_f16_pad<<<(total + 255) / 256, 256, 0, stream>>>(src, dst, sr, sc, stride, total);
  };
  cvt(w_ih_l0,  wih0f,  768, 432, 448, 768);
  cvt(w_ih_l0r, wih0rf, 768, 432, 448, 768);
  cvt(w_hh_l0,  whh0f,  768, 256, 256, 768);
  cvt(w_hh_l0r, whh0rf, 768, 256, 256, 768);
  cvt(w_ih_l1,  wih1f,  768, 512, 512, 768);
  cvt(w_ih_l1r, wih1rf, 768, 512, 512, 768);
  cvt(w_hh_l1,  whh1f,  768, 256, 256, 768);
  cvt(w_hh_l1r, whh1rf, 768, 256, 256, 768);
  cvt(fc_w,     fcwf,    62, 512, 512, 64);

  // ---- layer 0: input projections (WMMA) then persistent recurrence
  dim3 gGrid(768 / 64, M / 128);
  gemm_wmma_bias<<<gGrid, 256, 0, stream>>>(xf16, wih0f,  b_ih_l0,  xgF, 768, 448);
  gemm_wmma_bias<<<gGrid, 256, 0, stream>>>(xf16, wih0rf, b_ih_l0r, xgR, 768, 448);
  gru_layer<<<2, 1024, 0, stream>>>(xgF, xgR, whh0f, whh0rf, b_hh_l0, b_hh_l0r, h0f16, kT);

  // ---- layer 1
  gemm_wmma_bias<<<gGrid, 256, 0, stream>>>(h0f16, wih1f,  b_ih_l1,  xgF, 768, 512);
  gemm_wmma_bias<<<gGrid, 256, 0, stream>>>(h0f16, wih1rf, b_ih_l1r, xgR, 768, 512);
  gru_layer<<<2, 1024, 0, stream>>>(xgF, xgR, whh1f, whh1rf, b_hh_l1, b_hh_l1r, h1f16, kT);

  // ---- FC + mask
  fc_mask<<<M / 128, 256, 0, stream>>>(h1f16, fcwf, fc_b, lengths, out, kT);
}
